// QuantumCircuitLayer_76433237999992
// MI455X (gfx1250) — matrix operand order repaired and checked by reading, not compiled, not run on hardware
//
#include <hip/hip_runtime.h>

// ---------------------------------------------------------------------------
// Quantum state-vector simulator for MI455X (gfx1250).
//   - one workgroup (512 threads, 16 wave32) per batch sample
//   - full 2^16 state resident in LDS as packed half2 (re,im) = 256 KB
//   - XOR-linear bank swizzle -> inner-loop LDS addresses are base ^ const
//   - fused quad-wire 1q gates applied via V_WMMA_F32_16X16X32_F16
//     (32x32 realified tensor-product gate, f16 in / f32 accumulate)
//   - CNOT chain  = in-place LDS permutation
//   - CRZ gates   = one fused diagonal-phase pass per layer
//   - encoding RX + layer-0 1q gates folded into product-state init
// ---------------------------------------------------------------------------

typedef _Float16 f16;
typedef __attribute__((ext_vector_type(16))) _Float16     v16h;
typedef __attribute__((ext_vector_type(8)))  float        v8f;
typedef __attribute__((ext_vector_type(8)))  unsigned int v8u;

union ABReg { v16h h; v8u u; };

struct cf { float re, im; };
__device__ __forceinline__ cf cmul(cf a, cf b) {
  return { a.re*b.re - a.im*b.im, a.re*b.im + a.im*b.re };
}
__device__ __forceinline__ cf cadd(cf a, cf b) { return { a.re + b.re, a.im + b.im }; }

struct m2 { cf e[2][2]; };

// fused 1q gate: RX(vx) @ RZ(vz) @ RY(enc)
__device__ __forceinline__ m2 fused_gate(float enc, float vz, float vx) {
  float sy, cy, sz, cz, sx, cx;
  __sincosf(0.5f * enc, &sy, &cy);
  __sincosf(0.5f * vz,  &sz, &cz);
  __sincosf(0.5f * vx,  &sx, &cx);
  // RZ*RY : row0 scaled by e^{-i vz/2}, row1 by e^{+i vz/2}
  cf r00{ cy*cz, -cy*sz }, r01{ -sy*cz,  sy*sz };
  cf r10{ sy*cz,  sy*sz }, r11{  cy*cz,  cy*sz };
  // RX * M, RX = [[cx, -i sx], [-i sx, cx]] ; (-i s)*(a+bi) = s*b - i*s*a
  m2 g;
  g.e[0][0] = { cx*r00.re + sx*r10.im, cx*r00.im - sx*r10.re };
  g.e[0][1] = { cx*r01.re + sx*r11.im, cx*r01.im - sx*r11.re };
  g.e[1][0] = { cx*r10.re + sx*r00.im, cx*r10.im - sx*r00.re };
  g.e[1][1] = { cx*r11.re + sx*r01.im, cx*r11.im - sx*r01.re };
  return g;
}

__device__ __forceinline__ unsigned packc(cf a) {
  union { f16 h[2]; unsigned u; } t;
  t.h[0] = (f16)a.re; t.h[1] = (f16)a.im;
  return t.u;
}
__device__ __forceinline__ cf unpackc(unsigned u) {
  union { unsigned u; f16 h[2]; } t; t.u = u;
  return { (float)t.h[0], (float)t.h[1] };
}

// XOR-linear bank swizzle (dword units).  Injects a[11:8] -> addr[4:1] and
// a[15:12] -> addr[5:2].  Bijective on [0,65536); linear over GF(2), so
// swz(x ^ y) == swz(x) ^ swz(y)  -> inner-loop addresses fold to base ^ const.
__device__ constexpr unsigned swz(unsigned a) {
  return a ^ ((a >> 7) & 0x1eu) ^ ((a >> 10) & 0x3cu);
}
// insert 4-bit quad value q at bit position bp into 12-bit rest index
__device__ __forceinline__ unsigned amp_idx(unsigned rest, unsigned q, int bp) {
  unsigned lo = rest & ((1u << bp) - 1u);
  unsigned hi = rest >> bp;
  return (hi << (bp + 4)) | (q << bp) | lo;
}

#define MAT_OFF      262144u          // bytes: state = 65536 dwords
#define RED_OFF      270848u          // bytes: matbuf = 4*544 dwords (row pad 17)
#define SMEM_TOTAL   272896u          // bytes  (<= 320 KB WGP LDS)

// ---------------------------------------------------------------------------
// Precompute kernel: realified 32x32 f16 matrices for G(w)⊗G(w+1)⊗G(w+2)⊗G(w+3)
// for layers 1..3 (layer 0's 1q gates are folded into the init product state).
// Row-major M[n][k] == B-operand (B[k][n] = Mᵀ) column-major.
// ---------------------------------------------------------------------------
__global__ __launch_bounds__(256)
void build_quad_gates(const float* __restrict__ enc, const float* __restrict__ var,
                      f16* __restrict__ mats) {
  const int b = blockIdx.x;          // 0..11
  const int l = 1 + (b >> 2);        // layer 1..3
  const int t = b & 3;               // quad 0..3
  const int j = threadIdx.x;         // 0..255 -> complex entry (p,q) of 16x16
  const int p = j >> 4, q = j & 15;
  cf c{1.f, 0.f};
#pragma unroll
  for (int k2 = 0; k2 < 4; ++k2) {
    const int w  = 4*t + k2;
    const m2 g   = fused_gate(enc[l*16 + w], var[(l*16 + w)*3 + 0], var[(l*16 + w)*3 + 1]);
    const int sh = 3 - k2;           // wire 4t is most significant quad bit
    c = cmul(c, g.e[(p >> sh) & 1][(q >> sh) & 1]);
  }
  f16* M = mats + b * 1024;
  M[(2*p  )*32 + 2*q  ] = (f16)( c.re);
  M[(2*p  )*32 + 2*q+1] = (f16)(-c.im);
  M[(2*p+1)*32 + 2*q  ] = (f16)( c.im);
  M[(2*p+1)*32 + 2*q+1] = (f16)( c.re);
}

// ---------------------------------------------------------------------------
// One quad pass: apply 4 fused 1q gates (bits bp+3..bp) to the whole state.
// Templated on BP so all per-element address deltas become XOR constants.
// ---------------------------------------------------------------------------
template<int BP>
__device__ __forceinline__ void quad_pass(unsigned* state, const unsigned* mq,
                                          int wv, int lane) {
  const int n    = lane & 15;
  const int hi16 = lane >> 4;                         // 0 or 1
  // B operand (constant over the pass): lane n = column n_g, K-pairs per VGPR.
  // matbuf rows padded to 17 dwords -> conflict-free gather.
  ABReg B0, B1;
#pragma unroll
  for (int vv = 0; vv < 8; ++vv) {
    B0.u[vv] = mq[(n     ) * 17 + hi16 * 4 + (vv & 3) + (vv >> 2) * 8];
    B1.u[vv] = mq[(16 + n) * 17 + hi16 * 4 + (vv & 3) + (vv >> 2) * 8];
  }
  const unsigned q0 = (unsigned)(n >> 1);
  f16* sh = (f16*)state;
  for (int i = 0; i < 16; ++i) {
    const unsigned base = (unsigned)(wv * 256 + i * 16);
    // ---- A operand: each VGPR dword = one packed half2 amplitude -----------
    const unsigned rA = swz(amp_idx(base + (unsigned)n, (unsigned)(hi16 << 2), BP));
    ABReg A;
#pragma unroll
    for (int vv = 0; vv < 8; ++vv) {
      const unsigned qc = (unsigned)((vv & 3) | ((vv >> 2) << 3));  // compile-time
      A.u[vv] = state[rA ^ swz(qc << BP)];            // one XOR per load
    }
    const v8f z = {0.f, 0.f, 0.f, 0.f, 0.f, 0.f, 0.f, 0.f};
    v8f d0 = __builtin_amdgcn_wmma_f32_16x16x32_f16(false, A.h, false, B0.h,
                                                    (short)0, z, false, false);
    v8f d1 = __builtin_amdgcn_wmma_f32_16x16x32_f16(false, A.h, false, B1.h,
                                                    (short)0, z, false, false);
    // ---- D writeback: lane n owns components n and 16+n --------------------
    const unsigned w0 = swz(amp_idx(base + (unsigned)(hi16 * 8), q0, BP));
    const unsigned w1 = w0 ^ swz(8u << BP);
#pragma unroll
    for (int j = 0; j < 8; ++j) {
      const unsigned o = (BP == 0) ? ((unsigned)j << 4) : (unsigned)j;
      sh[2u * (w0 ^ o) + (unsigned)(n & 1)] = (f16)d0[j];
      sh[2u * (w1 ^ o) + (unsigned)(n & 1)] = (f16)d1[j];
    }
  }
}

// ---------------------------------------------------------------------------
// Main kernel: one block = one sample, state lives in LDS.
// ---------------------------------------------------------------------------
__global__ __launch_bounds__(512)
void qsim_kernel(const float* __restrict__ x, const float* __restrict__ enc,
                 const float* __restrict__ var, const float* __restrict__ ent,
                 const f16* __restrict__ mats, float* __restrict__ out) {
  extern __shared__ unsigned char smem[];
  unsigned* state  = (unsigned*)smem;                 // packed half2 amplitudes
  unsigned* matbuf = (unsigned*)(smem + MAT_OFF);     // 4 quad matrices, padded rows
  cf*       hibuf  = (cf*)(smem + RED_OFF);           // init scratch (2 KB)
  float*    red    = (float*)(smem + RED_OFF);        // reduction scratch (reuse)

  const int tid  = threadIdx.x;
  const int lane = tid & 31;
  const int wv   = tid >> 5;                          // wave 0..15
  const int s    = blockIdx.x;

  // ---- per-wire 2-vectors: v_w = FusedGate(l=0,w) * RX(x_w) * |0> ----------
  cf v[16][2];
#pragma unroll
  for (int w = 0; w < 16; ++w) {
    const m2 g = fused_gate(enc[w], var[w*3 + 0], var[w*3 + 1]);
    float sxh, cxh; __sincosf(0.5f * x[s*16 + w], &sxh, &cxh);
    const cf a{cxh, 0.f}, b{0.f, -sxh};               // RX|0> = (c, -i s)
    v[w][0] = cadd(cmul(g.e[0][0], a), cmul(g.e[0][1], b));
    v[w][1] = cadd(cmul(g.e[1][0], a), cmul(g.e[1][1], b));
  }

  // ---- product-state init: amp[a] = prod_w v_w[bit_{15-w}(a)] --------------
  if (tid < 256) {                                    // high-byte partial products
    cf p = v[0][(tid >> 7) & 1];
#pragma unroll
    for (int w = 1; w < 8; ++w) p = cmul(p, v[w][(tid >> (7 - w)) & 1]);
    hibuf[tid] = p;
  }
  __syncthreads();
  {
    const int lo8 = tid & 255;                        // low byte is constant/thread
    cf lp = v[8][(lo8 >> 7) & 1];
#pragma unroll
    for (int w = 9; w < 16; ++w) lp = cmul(lp, v[w][(lo8 >> (15 - w)) & 1]);
    for (int k = 0; k < 128; ++k) {
      const unsigned a = (unsigned)(k*512 + tid);
      state[swz(a)] = packc(cmul(hibuf[a >> 8], lp));
    }
  }
  __syncthreads();

  // ---- layers --------------------------------------------------------------
  for (int l = 0; l < 4; ++l) {
    if (l > 0) {
      // stage this layer's 4 quad matrices into LDS with 17-dword row pitch
      const unsigned* src = (const unsigned*)(mats + (l - 1) * 4096);
      for (int k = tid; k < 2048; k += 512) {
        const int q = k >> 9;                          // quad (512 dwords each)
        const int r = (k >> 4) & 31;                   // row within quad
        matbuf[q * 544 + r * 17 + (k & 15)] = src[k];
      }
      __syncthreads();

      quad_pass<12>(state, matbuf + 0 * 544, wv, lane); __syncthreads();
      quad_pass< 8>(state, matbuf + 1 * 544, wv, lane); __syncthreads();
      quad_pass< 4>(state, matbuf + 2 * 544, wv, lane); __syncthreads();
      quad_pass< 0>(state, matbuf + 3 * 544, wv, lane); __syncthreads();
    }

    // ---- CNOT chain: swap target pairs where control bit == 1 --------------
    for (int w = 0; w < 15; ++w) {
      const int pt = 14 - w;                           // target bit; control = pt+1
      const unsigned mask = (1u << pt) - 1u;
      const unsigned dsw  = swz(1u << pt);             // XOR delta target-bit flip
      for (int pi = tid; pi < 16384; pi += 512) {
        const unsigned lo = (unsigned)pi & mask;
        const unsigned hi = (unsigned)pi >> pt;
        const unsigned a0 = (hi << (pt + 2)) | (1u << (pt + 1)) | lo;
        const unsigned i0 = swz(a0), i1 = i0 ^ dsw;
        const unsigned t0 = state[i0], t1 = state[i1];
        state[i0] = t1; state[i1] = t0;
      }
      __syncthreads();
    }

    // ---- fused CRZ pass (all 4 controlled phases in one sweep) -------------
    {
      float cth[4], sth[4];
#pragma unroll
      for (int i = 0; i < 4; ++i)
        __sincosf(0.5f * ent[(l*4 + i) * 8], &sth[i], &cth[i]);
      for (int k = 0; k < 128; ++k) {
        const unsigned a = (unsigned)(k*512 + tid);
        cf ph{1.f, 0.f};
#pragma unroll
        for (int i = 0; i < 4; ++i) {
          if ((a >> (15 - i)) & 1u) {                  // control bit set
            const float sg = ((a >> (14 - i)) & 1u) ? sth[i] : -sth[i];
            ph = cmul(ph, cf{cth[i], sg});
          }
        }
        const unsigned idx = swz(a);
        state[idx] = packc(cmul(unpackc(state[idx]), ph));
      }
      __syncthreads();
    }
  }

  // ---- <Z> on wire 0 (MSB): sum |amp|^2 * (+1 / -1) ------------------------
  float acc = 0.f;
  for (int k = 0; k < 128; ++k) {
    const unsigned a = (unsigned)(k*512 + tid);
    const cf am = unpackc(state[swz(a)]);
    const float m = am.re*am.re + am.im*am.im;
    acc += ((a >> 15) & 1u) ? -m : m;
  }
  red[tid] = acc;
  __syncthreads();
  for (int off = 256; off > 0; off >>= 1) {
    if (tid < off) red[tid] += red[tid + off];
    __syncthreads();
  }
  if (tid == 0) out[s] = red[0];
}

// ---------------------------------------------------------------------------
extern "C" void kernel_launch(void* const* d_in, const int* in_sizes, int n_in,
                              void* d_out, int out_size, void* d_ws, size_t ws_size,
                              hipStream_t stream) {
  (void)n_in; (void)out_size; (void)ws_size;
  const float* x   = (const float*)d_in[0];   // (512, 16)
  const float* enc = (const float*)d_in[1];   // (4, 16)
  const float* var = (const float*)d_in[2];   // (4, 16, 3)
  const float* ent = (const float*)d_in[3];   // (4, 4, 8)
  f16*   mats = (f16*)d_ws;                   // 12 x 32x32 f16 = 24 KB
  float* out  = (float*)d_out;                // (512, 1)
  const int B = in_sizes[0] / 16;

  build_quad_gates<<<12, 256, 0, stream>>>(enc, var, mats);
  qsim_kernel<<<B, 512, SMEM_TOTAL, stream>>>(x, enc, var, ent, mats, out);
}
